// ExtractorHead_29326036697745
// MI455X (gfx1250) — matrix operand order repaired and checked by reading, not compiled, not run on hardware
//
#include <hip/hip_runtime.h>
#include <hip/hip_bf16.h>
#include <math.h>

// Problem constants (from reference setup_inputs)
#define NN 20000
#define EE 320000
#define HH 128
#define RR 20
#define PI_F 3.14159265358979323846f

typedef __bf16 bf16;
typedef __attribute__((ext_vector_type(16))) bf16 v16bf;
typedef __attribute__((ext_vector_type(8)))  float v8f;

__device__ __forceinline__ float silu_f(float x) {
  return x / (1.0f + __expf(-x));
}

// ---------------------------------------------------------------------------
// Weight packing: fp32 (K x Nout, row-major) -> bf16 in WMMA B-fragment order.
// Fragment layout (per 32x16 K-step x N-tile): 512 bf16.
//   lane l: n = ntile*16 + (l&15), kgrp = l>>4
//   vector element j (=2v+h): k = ks*32 + 8*kgrp + (j<8 ? j : j+8)
// So each lane's 16 bf16 are contiguous -> b128 loads in the GEMM.
// ---------------------------------------------------------------------------
__global__ void pack_w_kernel(const float* __restrict__ W, bf16* __restrict__ out,
                              int Kpad, int Kreal, int Nout) {
  int idx = blockIdx.x * blockDim.x + threadIdx.x;
  int total = Nout * Kpad;
  if (idx >= total) return;
  int frag   = idx >> 9;
  int within = idx & 511;
  int lane   = within >> 4;
  int j      = within & 15;
  int ksteps = Kpad >> 5;
  int ks = frag % ksteps;
  int nt = frag / ksteps;
  int n    = nt * 16 + (lane & 15);
  int kgrp = lane >> 4;
  int k = ks * 32 + 8 * kgrp + ((j < 8) ? j : (j + 8));
  float val = (k < Kreal) ? W[(size_t)k * Nout + n] : 0.0f;
  out[idx] = (bf16)val;
}

// ---------------------------------------------------------------------------
// Edge geometry: diff, dist, unit, cosine envelope, sinc RBF (bf16, padded 20->32)
// edist written straight into the output tail.
// ---------------------------------------------------------------------------
__global__ void edge_geom_kernel(const float* __restrict__ pos, const int* __restrict__ eidx,
                                 float* __restrict__ unitv, float* __restrict__ env,
                                 bf16* __restrict__ rbf, float* __restrict__ edist_out) {
  int e = blockIdx.x * blockDim.x + threadIdx.x;
  if (e >= EE) return;
  int r = eidx[e];        // receiver (edge[:,0])
  int s = eidx[EE + e];   // sender   (edge[:,1])
  float dx = pos[r * 3 + 0] - pos[s * 3 + 0];
  float dy = pos[r * 3 + 1] - pos[s * 3 + 1];
  float dz = pos[r * 3 + 2] - pos[s * 3 + 2];
  float d = sqrtf(dx * dx + dy * dy + dz * dz);
  float inv = 1.0f / d;
  unitv[(size_t)e * 3 + 0] = dx * inv;
  unitv[(size_t)e * 3 + 1] = dy * inv;
  unitv[(size_t)e * 3 + 2] = dz * inv;
  edist_out[e] = d;
  const float kf = PI_F / 6.0f;
  env[e] = (d < 6.0f) ? 0.5f * (__cosf(d * kf) + 1.0f) : 0.0f;
  bf16* rrow = rbf + (size_t)e * 32;
#pragma unroll
  for (int n = 0; n < RR; ++n)
    rrow[n] = (bf16)(__sinf(d * kf * (float)(n + 1)) * inv);
#pragma unroll
  for (int n = RR; n < 32; ++n)
    rrow[n] = (bf16)0.0f;
}

__global__ void embed_kernel(const int* __restrict__ z, const float* __restrict__ emb,
                             float* __restrict__ ns) {
  int idx = blockIdx.x * blockDim.x + threadIdx.x;
  if (idx >= NN * HH) return;
  int n = idx >> 7, h = idx & 127;
  ns[idx] = emb[(size_t)z[n] * HH + h];
}

// pos + edge_index (cast to float) into output tail
__global__ void tail_kernel(const float* __restrict__ pos, const int* __restrict__ eidx,
                            float* __restrict__ out_tail) {
  int idx = blockIdx.x * blockDim.x + threadIdx.x;
  const int npos = 3 * NN;
  const int ntot = npos + 2 * EE;
  if (idx >= ntot) return;
  out_tail[idx] = (idx < npos) ? pos[idx] : (float)eidx[idx - npos];
}

// ---------------------------------------------------------------------------
// Generic WMMA GEMM:  C[M,Nout] = act( bf16(A[M,K]) @ Wp + bias )
// grid = (ceil(Mtiles/4), Nout/16), block = 128 (4 waves).
// Tiling is 64 rows x 16 cols per block: each wave owns a distinct 16-row
// M-tile (streamed operand A is loaded exactly once); the small weight B
// fragment is re-read per wave but is always L2/WGP$-resident.
// A: fp32 row-major (converted per-fragment); Wp: packed bf16 (pack_w_kernel).
// ---------------------------------------------------------------------------
__global__ __launch_bounds__(128)
void gemm_kernel(const float* __restrict__ A, int lda,
                 const bf16* __restrict__ Wp, const float* __restrict__ bias,
                 float* __restrict__ C, int ldc, int K, int Mtiles, int act) {
  const int lane = threadIdx.x & 31;
  const int wave = threadIdx.x >> 5;
  const int mtile = blockIdx.x * 4 + wave;
  if (mtile >= Mtiles) return;  // wave-uniform exit: EXEC stays all-1 for live waves
  const int m0 = mtile * 16;
  const int ntile = blockIdx.y;
  const int n0 = ntile * 16;
  const int kgrp = lane >> 4;
  const int row = m0 + (lane & 15);
  const int ksteps = K >> 5;
  v8f acc = {};
  const bf16* wbase = Wp + (size_t)ntile * ksteps * 512 + lane * 16;
  for (int ks = 0; ks < ksteps; ++ks) {
    // A fragment (ISA 16-bit A layout): elem j=2v+h -> k = (v<4?2v:2v+8)+8*kgrp+h
    v16bf a;
    const float* arow = A + (size_t)row * lda + ks * 32 + kgrp * 8;
#pragma unroll
    for (int v = 0; v < 8; ++v) {
      int k0 = (v < 4) ? (2 * v) : (2 * v + 8);
      float2 t = *(const float2*)(arow + k0);
      a[2 * v]     = (bf16)t.x;
      a[2 * v + 1] = (bf16)t.y;
    }
    v16bf b = *(const v16bf*)(wbase + (size_t)ks * 512);
    if (ks + 1 < ksteps)
      __builtin_prefetch(wbase + (size_t)(ks + 1) * 512, 0, 3);
    acc = __builtin_amdgcn_wmma_f32_16x16x32_bf16(false, a, false, b, (short)0, acc,
                                                  false, false);
  }
  const int n = n0 + (lane & 15);
  float bv = bias ? bias[n] : 0.0f;
#pragma unroll
  for (int v = 0; v < 8; ++v) {
    int m = m0 + v + 8 * kgrp;  // C/D layout: vgpr v, lanes 16-31 -> M+8
    float x = acc[v] + bv;
    if (act) x = silu_f(x);
    C[(size_t)m * ldc + n] = x;
  }
}

// ---------------------------------------------------------------------------
// Fused message kernel, one block (4 waves) per 16-edge tile:
//   phase 1: filt = (rbf16x32 @ fW32x384 + fb) * env * scal[sender]  via WMMA,
//            spilled to LDS (fo[16][384]) -- never touches HBM.
//   phase 2: gated equivariant message + atomic scatter into ns / nv_new.
// Reads nv_old (pre-message), accumulates into nv_new (double buffered).
// ---------------------------------------------------------------------------
__global__ __launch_bounds__(128)
void message_kernel(const bf16* __restrict__ rbf, const float* __restrict__ env,
                    const float* __restrict__ unitv,
                    const int* __restrict__ erow, const int* __restrict__ ecol,
                    const float* __restrict__ scal,
                    const bf16* __restrict__ WfW, const float* __restrict__ fb,
                    const float* __restrict__ nv_old, float* __restrict__ nv_new,
                    float* __restrict__ ns) {
  __shared__ float fo[16 * 384];
  __shared__ int   s_send[16];
  __shared__ int   s_recv[16];
  __shared__ float s_env[16];

  const int tile = blockIdx.x;
  const int tid = threadIdx.x;
  const int lane = tid & 31;
  const int wave = tid >> 5;

  if (tid < 16) {
    int e = tile * 16 + tid;
    s_recv[tid] = erow[e];
    s_send[tid] = ecol[e];
    s_env[tid]  = env[e];
  }
  __syncthreads();

  // A fragment: 16 edges x 32 K of bf16 rbf (contiguous 1KB tile -> cache-hot
  // across the 4 waves of this block)
  const int kgrp = lane >> 4;
  const int mrow = lane & 15;
  const bf16* rrow = rbf + ((size_t)(tile * 16 + mrow)) * 32;
  v16bf a;
#pragma unroll
  for (int v = 0; v < 8; ++v) {
    int k0 = ((v < 4) ? (2 * v) : (2 * v + 8)) + 8 * kgrp;
    a[2 * v]     = rrow[k0];
    a[2 * v + 1] = rrow[k0 + 1];
  }

#pragma unroll
  for (int nt = 0; nt < 6; ++nt) {
    int ntile = wave * 6 + nt;                       // 24 N-tiles of 16 -> 384 cols
    v16bf b = *(const v16bf*)(WfW + (size_t)ntile * 512 + lane * 16);
    v8f acc = {};
    acc = __builtin_amdgcn_wmma_f32_16x16x32_bf16(false, a, false, b, (short)0, acc,
                                                  false, false);
    int n = ntile * 16 + (lane & 15);
    float bv = fb[n];
#pragma unroll
    for (int v = 0; v < 8; ++v) {
      int m = v + 8 * kgrp;
      float f = (acc[v] + bv) * s_env[m] * scal[(size_t)s_send[m] * 384 + n];
      fo[m * 384 + n] = f;
    }
  }
  __syncthreads();

  // phase 2: h = tid (blockDim == HH == 128)
  const int h = tid;
  for (int m = 0; m < 16; ++m) {
    int eg = tile * 16 + m;
    int r = s_recv[m], s = s_send[m];
    float gsv  = fo[m * 384 + h];
    float gev  = fo[m * 384 + 128 + h];
    float msgs = fo[m * 384 + 256 + h];
    atomicAdd(&ns[(size_t)r * HH + h], msgs);
#pragma unroll
    for (int d = 0; d < 3; ++d) {
      float mv = nv_old[((size_t)s * 3 + d) * HH + h] * gsv +
                 unitv[(size_t)eg * 3 + d] * gev;
      atomicAdd(&nv_new[((size_t)r * 3 + d) * HH + h], mv);
    }
  }
}

// ---------------------------------------------------------------------------
// Update helpers
// ---------------------------------------------------------------------------
__global__ void vn_concat_kernel(const float* __restrict__ Vv, const float* __restrict__ ns,
                                 float* __restrict__ X) {
  int idx = blockIdx.x * blockDim.x + threadIdx.x;
  if (idx >= NN * HH) return;
  int n = idx >> 7, h = idx & 127;
  float s = 1e-8f;
#pragma unroll
  for (int d = 0; d < 3; ++d) {
    float t = Vv[((size_t)n * 3 + d) * HH + h];
    s += t * t;
  }
  X[(size_t)n * 256 + h]       = sqrtf(s);
  X[(size_t)n * 256 + 128 + h] = ns[idx];
}

__global__ void update_final_kernel(const float* __restrict__ Uv, const float* __restrict__ Vv,
                                    const float* __restrict__ m, float* __restrict__ ns,
                                    float* __restrict__ nv) {
  int idx = blockIdx.x * blockDim.x + threadIdx.x;
  if (idx >= NN * HH) return;
  int n = idx >> 7, h = idx & 127;
  float inner = 0.0f;
#pragma unroll
  for (int d = 0; d < 3; ++d)
    inner += Uv[((size_t)n * 3 + d) * HH + h] * Vv[((size_t)n * 3 + d) * HH + h];
  float ass = m[(size_t)n * 384 + h];
  float asv = m[(size_t)n * 384 + 128 + h];
  float avv = m[(size_t)n * 384 + 256 + h];
  ns[idx] += ass + asv * inner;
#pragma unroll
  for (int d = 0; d < 3; ++d)
    nv[((size_t)n * 3 + d) * HH + h] += avv * Uv[((size_t)n * 3 + d) * HH + h];
}

// ---------------------------------------------------------------------------
extern "C" void kernel_launch(void* const* d_in, const int* in_sizes, int n_in,
                              void* d_out, int out_size, void* d_ws, size_t ws_size,
                              hipStream_t stream) {
  (void)in_sizes; (void)n_in; (void)out_size; (void)ws_size;
  const int*   z       = (const int*)d_in[0];
  const float* pos     = (const float*)d_in[1];
  const int*   eidx    = (const int*)d_in[2];
  const float* emb     = (const float*)d_in[3];
  const float* msg_fW  = (const float*)d_in[4];
  const float* msg_fb  = (const float*)d_in[5];
  const float* msg_w1  = (const float*)d_in[6];
  const float* msg_b1  = (const float*)d_in[7];
  const float* msg_w2  = (const float*)d_in[8];
  const float* msg_b2  = (const float*)d_in[9];
  const float* upd_U   = (const float*)d_in[10];
  const float* upd_V   = (const float*)d_in[11];
  const float* upd_w1  = (const float*)d_in[12];
  const float* upd_b1  = (const float*)d_in[13];
  const float* upd_w2  = (const float*)d_in[14];
  const float* upd_b2  = (const float*)d_in[15];
  const float* head_w1 = (const float*)d_in[16];
  const float* head_b1 = (const float*)d_in[17];
  const float* head_w2 = (const float*)d_in[18];
  const float* head_b2 = (const float*)d_in[19];

  // workspace carve-out (256B aligned)
  char* ws = (char*)d_ws;
  size_t off = 0;
  auto take = [&](size_t bytes) -> char* {
    char* p = ws + off;
    off = (off + bytes + 255) & ~(size_t)255;
    return p;
  };
  const size_t NSB = (size_t)NN * HH * 4;
  const size_t NVB = (size_t)NN * 3 * HH * 4;
  bf16*  Wp    = (bf16*)take(610304 * sizeof(bf16));
  float* ns    = (float*)take(NSB);
  float* nvA   = (float*)take(NVB);
  float* nvB   = (float*)take(NVB);
  float* unitv = (float*)take((size_t)EE * 3 * 4);
  float* env   = (float*)take((size_t)EE * 4);
  bf16*  rbf   = (bf16*)take((size_t)EE * 32 * sizeof(bf16));
  float* scal  = (float*)take((size_t)NN * 384 * 4);   // reused as update "m"
  float* tmp1  = (float*)take(NSB);
  float* X     = (float*)take((size_t)NN * 256 * 4);
  float* Uv    = (float*)take(NVB);
  float* Vv    = (float*)take(NVB);

  auto packLaunch = [&](const float* W, bf16* out, int Kpad, int Kreal, int Nout) {
    int total = Nout * Kpad;
    pack_w_kernel<<<(total + 255) / 256, 256, 0, stream>>>(W, out, Kpad, Kreal, Nout);
  };
  // per-round packed-weight block offsets (bf16 elements)
  const size_t PR = 192512;
  for (int r = 0; r < 3; ++r) {
    bf16* base = Wp + (size_t)r * PR;
    packLaunch(msg_fW + (size_t)r * RR * 384,  base + 0,      32, RR, 384);
    packLaunch(msg_w1 + (size_t)r * HH * HH,   base + 12288,  128, 128, 128);
    packLaunch(msg_w2 + (size_t)r * HH * 384,  base + 28672,  128, 128, 384);
    packLaunch(upd_U  + (size_t)r * HH * HH,   base + 77824,  128, 128, 128);
    packLaunch(upd_V  + (size_t)r * HH * HH,   base + 94208,  128, 128, 128);
    packLaunch(upd_w1 + (size_t)r * 2 * HH * HH, base + 110592, 256, 256, 128);
    packLaunch(upd_w2 + (size_t)r * HH * 384,  base + 143360, 128, 128, 384);
  }
  packLaunch(head_w1, Wp + 3 * PR,          128, 128, 128);
  packLaunch(head_w2, Wp + 3 * PR + 16384,  128, 128, 128);

  // output layout: v [N*H] | pos [3N] | edge_index [2E] | edist [E]
  float* out_v     = (float*)d_out;
  float* out_tail  = out_v + (size_t)NN * HH;
  float* out_edist = out_tail + 3 * NN + 2 * EE;

  edge_geom_kernel<<<(EE + 255) / 256, 256, 0, stream>>>(pos, eidx, unitv, env, rbf, out_edist);
  embed_kernel<<<(NN * HH + 255) / 256, 256, 0, stream>>>(z, emb, ns);
  tail_kernel<<<(3 * NN + 2 * EE + 255) / 256, 256, 0, stream>>>(pos, eidx, out_tail);
  hipMemsetAsync(nvA, 0, NVB, stream);

  // GEMM launch helper: M-tiles (rows/16) along x, N-tiles (cols/16) along y
  const int MT_N  = NN / 16;       // 1250
  const int MT_3N = 3 * NN / 16;   // 3750
  auto gemmLaunch = [&](const float* A, int lda, const bf16* W, const float* bias,
                        float* C, int ldc, int K, int Mtiles, int Nout, int act) {
    gemm_kernel<<<dim3((Mtiles + 3) / 4, Nout / 16), 128, 0, stream>>>(
        A, lda, W, bias, C, ldc, K, Mtiles, act);
  };

  float* nv_cur = nvA;
  float* nv_nxt = nvB;
  for (int r = 0; r < 3; ++r) {
    bf16* base = Wp + (size_t)r * PR;
    // scal = silu(ns@w1+b1)@w2+b2   (uses pre-message ns)
    gemmLaunch(ns, HH, base + 12288, msg_b1 + (size_t)r * HH, tmp1, HH, 128, MT_N, 128, 1);
    gemmLaunch(tmp1, HH, base + 28672, msg_b2 + (size_t)r * 384, scal, 384, 128, MT_N, 384, 0);
    // double-buffer nv: new = old + scatter(messages)
    hipMemcpyAsync(nv_nxt, nv_cur, NVB, hipMemcpyDeviceToDevice, stream);
    message_kernel<<<EE / 16, 128, 0, stream>>>(rbf, env, unitv, eidx, eidx + EE, scal,
        base + 0, msg_fb + (size_t)r * 384, nv_cur, nv_nxt, ns);
    // update: Uv/Vv as (3N x 128)x(128x128) WMMA GEMMs on post-message nv
    gemmLaunch(nv_nxt, HH, base + 77824, nullptr, Uv, HH, 128, MT_3N, 128, 0);
    gemmLaunch(nv_nxt, HH, base + 94208, nullptr, Vv, HH, 128, MT_3N, 128, 0);
    vn_concat_kernel<<<(NN * HH + 255) / 256, 256, 0, stream>>>(Vv, ns, X);
    gemmLaunch(X, 256, base + 110592, upd_b1 + (size_t)r * HH, tmp1, HH, 256, MT_N, 128, 1);
    gemmLaunch(tmp1, HH, base + 143360, upd_b2 + (size_t)r * 384, scal, 384, 128, MT_N, 384, 0);
    update_final_kernel<<<(NN * HH + 255) / 256, 256, 0, stream>>>(Uv, Vv, scal, ns, nv_nxt);
    float* t = nv_cur; nv_cur = nv_nxt; nv_nxt = t;
  }
  // head MLP -> d_out[0 : N*H]
  gemmLaunch(ns, HH, Wp + 3 * PR, head_b1, tmp1, HH, 128, MT_N, 128, 1);
  gemmLaunch(tmp1, HH, Wp + 3 * PR + 16384, head_b2, out_v, HH, 128, MT_N, 128, 0);
}
